// HybridQuanvolutionNet_65481071399322
// MI455X (gfx1250) — matrix operand order repaired and checked by reading, not compiled, not run on hardware
//
#include <hip/hip_runtime.h>
#include <hip/hip_bf16.h>
#include <math.h>

typedef __bf16 bf16;
typedef __attribute__((ext_vector_type(16))) __bf16 v16bf;
typedef __attribute__((ext_vector_type(8)))  __bf16 v8bf;
typedef __attribute__((ext_vector_type(8)))  float  v8f;

#define BATCH 128
#define XW    249        // input spatial
#define C1POOL 123       // pooled conv1 spatial
#define C2POOL 61        // pooled conv2 spatial
#define K1    55815      // 15*61*61, fc1 reduction dim
#define K1P   55840      // padded to multiple of 32
#define N1    120
#define N2    84

// ---------------------------------------------------------------- utilities
__global__ void fill_zero_u32(unsigned int* __restrict__ p, int nwords) {
  int t = blockIdx.x * blockDim.x + threadIdx.x;
  if (t < nwords) p[t] = 0u;
}

// ------------------------------------------------- conv1 + relu + pool (fused)
__global__ void conv1_pool_kernel(const float* __restrict__ x,
                                  const float* __restrict__ w,
                                  const float* __restrict__ b,
                                  float* __restrict__ h1) {
  int t = blockIdx.x * blockDim.x + threadIdx.x;
  const int total = BATCH * 6 * C1POOL * C1POOL;
  if (t >= total) return;
  int px = t % C1POOL; int r = t / C1POOL;
  int py = r % C1POOL; r /= C1POOL;
  int co = r % 6;      int bt = r / 6;
  const float* wk = w + co * 3 * 25;
  float best = -1e30f;
  for (int dy = 0; dy < 2; ++dy)
  for (int dx = 0; dx < 2; ++dx) {
    int cy = py + dy, cx = px + dx;   // conv-out coords, < 124
    float acc = b[co];
    for (int ci = 0; ci < 3; ++ci) {
      const float* xp = x + ((size_t)bt * 3 + ci) * XW * XW;
      const float* wp = wk + ci * 25;
      for (int ky = 0; ky < 5; ++ky) {
        int iy = cy * 2 - 1 + ky;
        if (iy < 0 || iy >= XW) continue;
        for (int kx = 0; kx < 5; ++kx) {
          int ix = cx * 2 - 1 + kx;
          if (ix < 0 || ix >= XW) continue;
          acc += xp[iy * XW + ix] * wp[ky * 5 + kx];
        }
      }
    }
    acc = acc > 0.f ? acc : 0.f;      // relu, then pool-max
    best = fmaxf(best, acc);
  }
  h1[t] = best;
}

// ----------------------------- conv2 + relu + pool, write bf16 into padded h2
__global__ void conv2_pool_kernel(const float* __restrict__ h1,
                                  const float* __restrict__ w,
                                  const float* __restrict__ b,
                                  bf16* __restrict__ h2bf) {
  int t = blockIdx.x * blockDim.x + threadIdx.x;
  const int total = BATCH * 15 * C2POOL * C2POOL;
  if (t >= total) return;
  int px = t % C2POOL; int r = t / C2POOL;
  int py = r % C2POOL; r /= C2POOL;
  int co = r % 15;     int bt = r / 15;
  float best = -1e30f;
  for (int dy = 0; dy < 2; ++dy)
  for (int dx = 0; dx < 2; ++dx) {
    int cy = py + dy, cx = px + dx;   // conv-out coords, < 62
    float acc = b[co];
    for (int ci = 0; ci < 6; ++ci) {
      const float* hp = h1 + ((size_t)bt * 6 + ci) * C1POOL * C1POOL;
      const float* wp = w + (co * 6 + ci) * 9;
      for (int ky = 0; ky < 3; ++ky) {
        int iy = cy * 2 - 1 + ky;
        if (iy < 0 || iy >= C1POOL) continue;
        for (int kx = 0; kx < 3; ++kx) {
          int ix = cx * 2 - 1 + kx;
          if (ix < 0 || ix >= C1POOL) continue;
          acc += hp[iy * C1POOL + ix] * wp[ky * 3 + kx];
        }
      }
    }
    acc = acc > 0.f ? acc : 0.f;
    best = fmaxf(best, acc);
  }
  // NCHW flatten index == co*3721 + py*61 + px, stored in K1P-padded row
  h2bf[(size_t)bt * K1P + co * (C2POOL * C2POOL) + py * C2POOL + px] = (bf16)best;
}

// ------------------------------------------ fc1 weight fp32 -> padded bf16
__global__ void convert_w1_kernel(const float* __restrict__ w, bf16* __restrict__ wb) {
  int t = blockIdx.x * blockDim.x + threadIdx.x;
  const int total = N1 * K1;
  if (t >= total) return;
  int n = t / K1, k = t % K1;
  wb[(size_t)n * K1P + k] = (bf16)w[t];
}

// ------------------------------------------------------- fc1 via bf16 WMMA
// D(128x120) = A(128xK1P bf16) * B^T(rows of fc1_w, bf16), f32 accumulate.
__global__ void __launch_bounds__(32)
fc1_wmma_kernel(const bf16* __restrict__ A, const bf16* __restrict__ Bw,
                const float* __restrict__ bias, float* __restrict__ out) {
  const int lane = threadIdx.x & 31;
  const int m0 = blockIdx.x * 16;
  const int n0 = blockIdx.y * 16;
  const int ln = lane & 15;
  const int kh = lane >> 4;
  const bf16* arow = A  + (size_t)(m0 + ln) * K1P;
  const bf16* brow = Bw + (size_t)(n0 + ln) * K1P;
  v8f acc = {};
  for (int k = 0; k < K1P; k += 32) {
    union { v16bf v; v8bf h[2]; } a;
    a.h[0] = *(const v8bf*)(arow + k + kh * 16);       // K = k+kh*16 .. +7
    a.h[1] = *(const v8bf*)(arow + k + kh * 16 + 8);   // K = k+kh*16+8 .. +15
    v16bf bvec = *(const v16bf*)(brow + k + kh * 16);  // K = k+kh*16 .. +15
    __builtin_prefetch(brow + k + 512, 0, 1);          // global_prefetch_b8
    acc = __builtin_amdgcn_wmma_f32_16x16x32_bf16(
        false, a.v, false, bvec, (short)0, acc, false, false);
  }
  const int col = n0 + ln;
  if (col < N1) {
    float bb = bias[col];
#pragma unroll
    for (int r = 0; r < 8; ++r) {
      int row = m0 + kh * 8 + r;
      float v = acc[r] + bb;
      out[row * N1 + col] = v > 0.f ? v : 0.f;
    }
  }
}

// ---------------------------------------------------------------- fc2 + relu
__global__ void fc2_kernel(const float* __restrict__ in, const float* __restrict__ w,
                           const float* __restrict__ b, float* __restrict__ out) {
  int t = blockIdx.x * blockDim.x + threadIdx.x;
  if (t >= BATCH * N2) return;
  int bt = t / N2, o = t % N2;
  float acc = b[o];
  const float* ip = in + bt * N1;
  const float* wp = w + o * N1;
  for (int k = 0; k < N1; ++k) acc += ip[k] * wp[k];
  out[t] = acc > 0.f ? acc : 0.f;
}

// ------------------------------------------------- 4-qubit circuit helpers
// state index bits: wire w -> bit (3-w) (wire 0 is MSB of flattened (2,2,2,2))
__device__ __forceinline__ void ry_gate(float* s, float th, int w) {
  float c = cosf(0.5f * th), sn = sinf(0.5f * th);
  int mask = 1 << (3 - w);
  for (int i = 0; i < 16; ++i) {
    if (i & mask) continue;
    int j = i | mask;
    float a = s[i], b = s[j];
    s[i] = c * a - sn * b;
    s[j] = sn * a + c * b;
  }
}
__device__ __forceinline__ void cnot_gate(float* s, int ctrl, int tgt) {
  int cm = 1 << (3 - ctrl), tm = 1 << (3 - tgt);
  for (int i = 0; i < 16; ++i) {
    if ((i & cm) && !(i & tm)) {
      int j = i | tm;
      float tv = s[i]; s[i] = s[j]; s[j] = tv;
    }
  }
}
__device__ __forceinline__ void random_layer(float* s, const float* th) {
  for (int i = 0; i < 4; ++i) {
    ry_gate(s, th[i], i);
    cnot_gate(s, i, (i + 1) & 3);
  }
}
__device__ __forceinline__ float measz(const float* s, int w) {
  int mask = 1 << (3 - w);
  float p = 0.f;
  for (int i = 0; i < 16; ++i) p += (i & mask) ? -s[i] * s[i] : s[i] * s[i];
  return p;
}

// ------------------------------------------------ fc3 + quantum head -> prob
__global__ void fc3_head_kernel(const float* __restrict__ fc2o,
                                const float* __restrict__ w3, const float* __restrict__ b3,
                                const float* __restrict__ thetas,
                                float* __restrict__ pconv) {
  int bt = blockIdx.x * blockDim.x + threadIdx.x;
  if (bt >= BATCH) return;
  float logit = b3[0];
  const float* h = fc2o + bt * N2;
  for (int i = 0; i < N2; ++i) logit += h[i] * w3[i];
  float s[16];
  for (int i = 0; i < 16; ++i) s[i] = 0.f;
  s[0] = 1.f;
  ry_gate(s, logit, 0);
  random_layer(s, thetas);
  float z = measz(s, 0);
  pconv[bt] = 1.f / (1.f + expf(-z));
}

// ----------------------- separable antialiased linear resize (jax semantics)
// pass 1: rows 249 -> 28, gray = channel mean computed on the fly
__global__ void resize_rows_kernel(const float* __restrict__ x, float* __restrict__ gtmp) {
  int t = blockIdx.x * blockDim.x + threadIdx.x;
  const int total = BATCH * 28 * XW;
  if (t >= total) return;
  int cx = t % XW; int r = t / XW;
  int oy = r % 28; int bt = r / 28;
  const float inv = (float)XW / 28.0f;   // kernel radius (antialias)
  const float sc  = 28.0f / (float)XW;
  float cyc = (oy + 0.5f) * inv - 0.5f;
  int j0 = (int)ceilf(cyc - inv);  if (j0 < 0) j0 = 0;
  int j1 = (int)floorf(cyc + inv); if (j1 > XW - 1) j1 = XW - 1;
  const float* x0 = x + (size_t)bt * 3 * XW * XW;
  float acc = 0.f, wsum = 0.f;
  for (int j = j0; j <= j1; ++j) {
    float wgt = 1.0f - fabsf((float)j - cyc) * sc;
    if (wgt <= 0.f) continue;
    float gv = (x0[j * XW + cx] + x0[XW * XW + j * XW + cx] +
                x0[2 * XW * XW + j * XW + cx]) * (1.0f / 3.0f);
    acc += wgt * gv; wsum += wgt;
  }
  gtmp[t] = acc / wsum;
}
// pass 2: cols 249 -> 28
__global__ void resize_cols_kernel(const float* __restrict__ gtmp, float* __restrict__ g) {
  int t = blockIdx.x * blockDim.x + threadIdx.x;
  const int total = BATCH * 28 * 28;
  if (t >= total) return;
  int ox = t % 28; int r = t / 28;
  int oy = r % 28; int bt = r / 28;
  const float inv = (float)XW / 28.0f;
  const float sc  = 28.0f / (float)XW;
  float cxc = (ox + 0.5f) * inv - 0.5f;
  int j0 = (int)ceilf(cxc - inv);  if (j0 < 0) j0 = 0;
  int j1 = (int)floorf(cxc + inv); if (j1 > XW - 1) j1 = XW - 1;
  const float* row = gtmp + ((size_t)bt * 28 + oy) * XW;
  float acc = 0.f, wsum = 0.f;
  for (int j = j0; j <= j1; ++j) {
    float wgt = 1.0f - fabsf((float)j - cxc) * sc;
    if (wgt <= 0.f) continue;
    acc += wgt * row[j]; wsum += wgt;
  }
  g[t] = acc / wsum;
}

// ------------------------------------------ quanvolution: one circuit/patch
__global__ void quanv_kernel(const float* __restrict__ g, const float* __restrict__ thetas,
                             float* __restrict__ feats) {
  int t = blockIdx.x * blockDim.x + threadIdx.x;
  if (t >= BATCH * 196) return;
  int bt = t / 196, p = t % 196;
  int py = p / 14, px = p % 14;
  float s[16];
  for (int i = 0; i < 16; ++i) s[i] = 0.f;
  s[0] = 1.f;
  for (int w = 0; w < 4; ++w) {
    float th = g[(size_t)bt * 784 + (2 * py + (w >> 1)) * 28 + (2 * px + (w & 1))];
    ry_gate(s, th, w);
  }
  random_layer(s, thetas);
  for (int w = 0; w < 4; ++w)
    feats[(size_t)bt * 784 + p * 4 + w] = measz(s, w);
}

// ---------------------------------------------------------------- final mix
__global__ void final_kernel(const float* __restrict__ feats, const float* __restrict__ lw,
                             const float* __restrict__ lb, const float* __restrict__ pconv,
                             float* __restrict__ out) {
  int bt = blockIdx.x * blockDim.x + threadIdx.x;
  if (bt >= BATCH) return;
  float acc = lb[0];
  const float* f = feats + (size_t)bt * 784;
  for (int i = 0; i < 784; ++i) acc += f[i] * lw[i];
  float pq = 1.f / (1.f + expf(-acc));
  float pr = 0.5f * (pconv[bt] + pq);
  out[bt * 2]     = pr;
  out[bt * 2 + 1] = 1.f - pr;
}

// ---------------------------------------------------------------------------
extern "C" void kernel_launch(void* const* d_in, const int* in_sizes, int n_in,
                              void* d_out, int out_size, void* d_ws, size_t ws_size,
                              hipStream_t stream) {
  (void)in_sizes; (void)n_in; (void)out_size; (void)ws_size;
  const float* x   = (const float*)d_in[0];
  const float* c1w = (const float*)d_in[1];
  const float* c1b = (const float*)d_in[2];
  const float* c2w = (const float*)d_in[3];
  const float* c2b = (const float*)d_in[4];
  const float* f1w = (const float*)d_in[5];
  const float* f1b = (const float*)d_in[6];
  const float* f2w = (const float*)d_in[7];
  const float* f2b = (const float*)d_in[8];
  const float* f3w = (const float*)d_in[9];
  const float* f3b = (const float*)d_in[10];
  const float* hth = (const float*)d_in[11];
  const float* qth = (const float*)d_in[12];
  const float* lw  = (const float*)d_in[13];
  const float* lb  = (const float*)d_in[14];
  float* out = (float*)d_out;

  char* ws = (char*)d_ws;
  size_t off = 0;
  auto alloc = [&](size_t bytes) -> char* {
    char* p = ws + off;
    off += (bytes + 255) & ~(size_t)255;
    return p;
  };
  const size_t H1_ELEMS = (size_t)BATCH * 6 * C1POOL * C1POOL;   // 11,619,072
  float* h1   = (float*)alloc(H1_ELEMS * 4);
  bf16*  h2bf = (bf16*) alloc((size_t)BATCH * K1P * 2);  // size % 256 == 0
  bf16*  w1bf = (bf16*) alloc((size_t)BATCH * K1P * 2);  // adjacent to h2bf
  float* fc1o = (float*)alloc((size_t)BATCH * N1 * 4);
  float* fc2o = (float*)alloc((size_t)BATCH * N2 * 4);
  float* pcv  = (float*)alloc((size_t)BATCH * 4);
  // Overlay resize/quanv scratch onto h1's region (h1 is dead after conv2;
  // single-stream ordering serializes the reuse).
  float* gtmp  = (float*)h1;                       // 128*28*249
  float* g28   = gtmp + (size_t)BATCH * 28 * XW;   // 128*28*28
  float* feats = g28 + (size_t)BATCH * 784;        // 128*784

  const int TPB = 256;
  // zero the padded bf16 staging buffers (h2bf + w1bf are contiguous)
  int fill_words = (int)((size_t)2 * BATCH * K1P * 2 / 4);
  fill_zero_u32<<<(fill_words + TPB - 1) / TPB, TPB, 0, stream>>>((unsigned int*)h2bf, fill_words);

  int n1 = BATCH * 6 * C1POOL * C1POOL;
  conv1_pool_kernel<<<(n1 + TPB - 1) / TPB, TPB, 0, stream>>>(x, c1w, c1b, h1);

  int n2 = BATCH * 15 * C2POOL * C2POOL;
  conv2_pool_kernel<<<(n2 + TPB - 1) / TPB, TPB, 0, stream>>>(h1, c2w, c2b, h2bf);

  int nw = N1 * K1;
  convert_w1_kernel<<<(nw + TPB - 1) / TPB, TPB, 0, stream>>>(f1w, w1bf);

  fc1_wmma_kernel<<<dim3(BATCH / 16, 128 / 16), 32, 0, stream>>>(h2bf, w1bf, f1b, fc1o);

  int nf2 = BATCH * N2;
  fc2_kernel<<<(nf2 + TPB - 1) / TPB, TPB, 0, stream>>>(fc1o, f2w, f2b, fc2o);

  fc3_head_kernel<<<1, BATCH, 0, stream>>>(fc2o, f3w, f3b, hth, pcv);

  int ng1 = BATCH * 28 * XW;
  resize_rows_kernel<<<(ng1 + TPB - 1) / TPB, TPB, 0, stream>>>(x, gtmp);
  int ng2 = BATCH * 28 * 28;
  resize_cols_kernel<<<(ng2 + TPB - 1) / TPB, TPB, 0, stream>>>(gtmp, g28);

  int nq = BATCH * 196;
  quanv_kernel<<<(nq + TPB - 1) / TPB, TPB, 0, stream>>>(g28, qth, feats);

  final_kernel<<<1, BATCH, 0, stream>>>(feats, lw, lb, pcv, out);
}
